// AttentionCell_32830730011135
// MI455X (gfx1250) — matrix operand order repaired and checked by reading, not compiled, not run on hardware
//
#include <hip/hip_runtime.h>

typedef __attribute__((ext_vector_type(16))) __bf16 v16bf;
typedef __attribute__((ext_vector_type(8)))  float  v8f;

// Async global->LDS builtin operand types (prototype revealed by round-2 diag:
// param0 = int __vector_size__(16) * in device (as1) address space).
typedef int vint4 __attribute__((__vector_size__(16)));
typedef __attribute__((address_space(1))) vint4* gas_v4i;
typedef __attribute__((address_space(3))) vint4* las_v4i;

#define Hd  512
#define Td  256
#define Bd  256
#define Ed  96

#if defined(__has_builtin)
#if __has_builtin(__builtin_amdgcn_global_load_async_to_lds_b128) && \
    __has_builtin(__builtin_amdgcn_s_wait_asynccnt)
#define USE_ASYNC_LDS 1
#endif
#endif

__device__ __forceinline__ float sigmoidf_(float x) { return 1.0f / (1.0f + expf(-x)); }

// ---------------------------------------------------------------------------
// Kernel 1: pre-swizzle W_i2h (f32 -> bf16) into WMMA B-fragment layout.
// B frag (32x16, 16-bit): element i of lane L holds K = 8*(L/16) + (i<8 ? i : i+8), N = L%16.
// Storage: [nt (32)][k (16)][lane (32)][i (16)] so each lane reads 32B contiguous.
// ---------------------------------------------------------------------------
__global__ void swizzle_W(const float* __restrict__ W, __bf16* __restrict__ Wswz) {
    int idx  = blockIdx.x * blockDim.x + threadIdx.x;    // 0 .. 262143
    int i    = idx & 15;
    int lane = (idx >> 4) & 31;
    int k    = (idx >> 9) & 15;
    int nt   = idx >> 13;
    int half = lane >> 4;
    int n    = nt * 16 + (lane & 15);
    int K    = 32 * k + 8 * half + (i < 8 ? i : i + 8);
    Wswz[idx] = (__bf16)W[K * Hd + n];
}

// ---------------------------------------------------------------------------
// Kernel 2: hp = prev_h @ W_h2h + b_h2h   [256,512]x[512,512] (tiny, f32 VALU)
// ---------------------------------------------------------------------------
__global__ void hp_kernel(const float* __restrict__ prev_h,
                          const float* __restrict__ W_h2h,
                          const float* __restrict__ b_h2h,
                          float* __restrict__ hp) {
    int idx = blockIdx.x * blockDim.x + threadIdx.x;     // 0 .. B*H-1
    int b = idx >> 9;
    int h = idx & (Hd - 1);
    float s = b_h2h[h];
    const float* pr = prev_h + b * Hd;
    for (int d = 0; d < Hd; ++d) s = fmaf(pr[d], W_h2h[d * Hd + h], s);
    hp[idx] = s;
}

// ---------------------------------------------------------------------------
// Kernel 3: attention scores via WMMA bf16, 8 waves / workgroup.
// Wave w owns M-tile (blockIdx*8+w); the 16KB B-block for each N-tile is
// staged once per workgroup in LDS (async global->LDS, double buffered) and
// shared by all 8 waves => 8x less L2 traffic on W.
//   acc(16x16,f32) = sum_k A_frag[k] * B_frag[nt,k]   (16 x wmma 16x16x32 bf16)
//   e_partial[row] += tanh(acc + hp[b,n]) * w_score[n]
// ---------------------------------------------------------------------------
__global__ void __launch_bounds__(256)
attn_scores(const float*  __restrict__ batch_H,
            const __bf16* __restrict__ Wswz,
            const float*  __restrict__ hp,
            const float*  __restrict__ w_score,
            float*        __restrict__ e) {
    __shared__ __align__(64) __bf16 Bbuf[2][16 * 32 * 16];   // 2 x 16 KB

    int tid   = threadIdx.x;
    int wave  = tid >> 5;
    int lane  = tid & 31;
    int mtile = blockIdx.x * 8 + wave;   // 0 .. 4095
    int half  = lane >> 4;
    int l15   = lane & 15;
    int b     = mtile >> 4;              // 16 M-tiles per batch row (T=256)

    // Load all A fragments for this wave's tile (16 K-steps of 32).
    // 16-bit A layout: lane (l15 = row M), element i -> K = 32k + 8*half + (i<8 ? i : i+8)
    v16bf afrag[16];
    const float* rowp = batch_H + (long)(mtile * 16 + l15) * Hd;
#pragma unroll
    for (int k = 0; k < 16; ++k) {
        int a = 32 * k + 8 * half;
        float4 p0 = *(const float4*)(rowp + a);
        float4 p1 = *(const float4*)(rowp + a + 4);
        float4 p2 = *(const float4*)(rowp + a + 16);
        float4 p3 = *(const float4*)(rowp + a + 20);
        v16bf f;
        f[0]  = (__bf16)p0.x; f[1]  = (__bf16)p0.y; f[2]  = (__bf16)p0.z; f[3]  = (__bf16)p0.w;
        f[4]  = (__bf16)p1.x; f[5]  = (__bf16)p1.y; f[6]  = (__bf16)p1.z; f[7]  = (__bf16)p1.w;
        f[8]  = (__bf16)p2.x; f[9]  = (__bf16)p2.y; f[10] = (__bf16)p2.z; f[11] = (__bf16)p2.w;
        f[12] = (__bf16)p3.x; f[13] = (__bf16)p3.y; f[14] = (__bf16)p3.z; f[15] = (__bf16)p3.w;
        afrag[k] = f;
    }

    float esum[8];
#pragma unroll
    for (int r = 0; r < 8; ++r) esum[r] = 0.0f;

#ifdef USE_ASYNC_LDS
    // Prologue: async-copy B tile 0 into buffer 0 (each thread moves 4x16B).
#pragma unroll
    for (int c = 0; c < 4; ++c)
        __builtin_amdgcn_global_load_async_to_lds_b128(
            (gas_v4i)(Wswz + tid * 8 + c * 2048),
            (las_v4i)(&Bbuf[0][tid * 8 + c * 2048]),
            0, 0);

    for (int nt = 0; nt < 32; ++nt) {
        int cur = nt & 1;
        __builtin_amdgcn_s_wait_asynccnt(0);   // my copies into Bbuf[cur] done
        __syncthreads();                       // everyone's copies visible
        if (nt + 1 < 32) {                     // prefetch next tile into other buffer
            const __bf16* src = Wswz + (nt + 1) * 8192 + tid * 8;
#pragma unroll
            for (int c = 0; c < 4; ++c)
                __builtin_amdgcn_global_load_async_to_lds_b128(
                    (gas_v4i)(src + c * 2048),
                    (las_v4i)(&Bbuf[cur ^ 1][tid * 8 + c * 2048]),
                    0, 0);
        }
        v8f acc = {};
        const __bf16* bp = &Bbuf[cur][lane * 16];
#pragma unroll
        for (int k = 0; k < 16; ++k) {
            v16bf bfrag = *(const v16bf*)(bp + k * 512);
            acc = __builtin_amdgcn_wmma_f32_16x16x32_bf16(
                false, afrag[k], false, bfrag, (short)0, acc, false, false);
        }
        // C/D layout: element r of lane -> M = r + 8*half, N = l15 (+16*nt)
        int   n   = nt * 16 + l15;
        float hpv = hp[b * Hd + n];
        float wsv = w_score[n];
#pragma unroll
        for (int r = 0; r < 8; ++r)
            esum[r] += tanhf(acc[r] + hpv) * wsv;
    }
#else
    for (int nt = 0; nt < 32; ++nt) {
        __syncthreads();                       // previous tile fully consumed
        const __bf16* src = Wswz + nt * 8192 + tid * 8;
#pragma unroll
        for (int c = 0; c < 4; ++c)
            *(int4*)&Bbuf[0][tid * 8 + c * 2048] = *(const int4*)(src + c * 2048);
        __syncthreads();

        v8f acc = {};
        const __bf16* bp = &Bbuf[0][lane * 16];
#pragma unroll
        for (int k = 0; k < 16; ++k) {
            v16bf bfrag = *(const v16bf*)(bp + k * 512);
            acc = __builtin_amdgcn_wmma_f32_16x16x32_bf16(
                false, afrag[k], false, bfrag, (short)0, acc, false, false);
        }
        int   n   = nt * 16 + l15;
        float hpv = hp[b * Hd + n];
        float wsv = w_score[n];
#pragma unroll
        for (int r = 0; r < 8; ++r)
            esum[r] += tanhf(acc[r] + hpv) * wsv;
    }
#endif

    // Reduce over the 16 lanes (N dimension) within each half-wave.
#pragma unroll
    for (int r = 0; r < 8; ++r) {
        float v = esum[r];
        v += __shfl_xor(v, 1, 32);
        v += __shfl_xor(v, 2, 32);
        v += __shfl_xor(v, 4, 32);
        v += __shfl_xor(v, 8, 32);
        esum[r] = v;
    }
    if (l15 == 0) {
        int base = mtile * 16 + 8 * half;   // lanes 0 / 16 write rows 0-7 / 8-15
#pragma unroll
        for (int r = 0; r < 8; ++r) e[base + r] = esum[r];
    }
}

// ---------------------------------------------------------------------------
// Kernel 4: softmax over T + context = alpha^T @ batch_H  (per-batch block)
// ---------------------------------------------------------------------------
__global__ void softmax_ctx(const float* __restrict__ e,
                            const float* __restrict__ batch_H,
                            float* __restrict__ alpha_out,
                            float* __restrict__ context) {
    __shared__ float sa[Td];
    __shared__ float red[256];
    int b = blockIdx.x;
    int t = threadIdx.x;

    float v = e[b * Td + t];
    red[t] = v;
    __syncthreads();
    for (int s = 128; s > 0; s >>= 1) {
        if (t < s) red[t] = fmaxf(red[t], red[t + s]);
        __syncthreads();
    }
    float mx = red[0];
    __syncthreads();
    float ex = expf(v - mx);
    red[t] = ex;
    __syncthreads();
    for (int s = 128; s > 0; s >>= 1) {
        if (t < s) red[t] += red[t + s];
        __syncthreads();
    }
    float al = ex / red[0];
    sa[t] = al;
    alpha_out[b * Td + t] = al;
    __syncthreads();

    const float* bh = batch_H + (long)b * Td * Hd;
    for (int d = t; d < Hd; d += 256) {
        float s = 0.0f;
        for (int tt = 0; tt < Td; ++tt) s = fmaf(sa[tt], bh[tt * Hd + d], s);
        context[b * Hd + d] = s;
    }
}

// ---------------------------------------------------------------------------
// Kernel 5: LSTM cell (f32):  z = [context|onehot]@W_lstm + prev_h@U_lstm + b
// Gate order i, f, g, o.  One block per batch row, one thread per hidden unit.
// ---------------------------------------------------------------------------
__global__ void lstm_kernel(const float* __restrict__ prev_h,
                            const float* __restrict__ prev_c,
                            const float* __restrict__ onehot,
                            const float* __restrict__ context,
                            const float* __restrict__ W_lstm,
                            const float* __restrict__ U_lstm,
                            const float* __restrict__ b_lstm,
                            float* __restrict__ out_h,
                            float* __restrict__ out_c) {
    __shared__ float xs[Hd + Ed];
    __shared__ float hs[Hd];
    int b   = blockIdx.x;
    int tid = threadIdx.x;           // 0 .. 511
    xs[tid] = context[b * Hd + tid];
    hs[tid] = prev_h[b * Hd + tid];
    if (tid < Ed) xs[Hd + tid] = onehot[b * Ed + tid];
    __syncthreads();

    int h = tid;
    float zi = b_lstm[h];
    float zf = b_lstm[Hd + h];
    float zg = b_lstm[2 * Hd + h];
    float zo = b_lstm[3 * Hd + h];
    for (int k = 0; k < Hd + Ed; ++k) {
        float xv = xs[k];
        const float* row = W_lstm + (long)k * 4 * Hd;
        zi = fmaf(xv, row[h], zi);
        zf = fmaf(xv, row[Hd + h], zf);
        zg = fmaf(xv, row[2 * Hd + h], zg);
        zo = fmaf(xv, row[3 * Hd + h], zo);
    }
    for (int k = 0; k < Hd; ++k) {
        float hv = hs[k];
        const float* row = U_lstm + (long)k * 4 * Hd;
        zi = fmaf(hv, row[h], zi);
        zf = fmaf(hv, row[Hd + h], zf);
        zg = fmaf(hv, row[2 * Hd + h], zg);
        zo = fmaf(hv, row[3 * Hd + h], zo);
    }
    float cn = sigmoidf_(zf) * prev_c[b * Hd + h] + sigmoidf_(zi) * tanhf(zg);
    float hn = sigmoidf_(zo) * tanhf(cn);
    out_h[b * Hd + h] = hn;
    out_c[b * Hd + h] = cn;
}

// ---------------------------------------------------------------------------
extern "C" void kernel_launch(void* const* d_in, const int* in_sizes, int n_in,
                              void* d_out, int out_size, void* d_ws, size_t ws_size,
                              hipStream_t stream) {
    const float* prev_h  = (const float*)d_in[0];
    const float* prev_c  = (const float*)d_in[1];
    const float* batch_H = (const float*)d_in[2];
    const float* onehot  = (const float*)d_in[3];
    const float* W_i2h   = (const float*)d_in[4];
    const float* W_h2h   = (const float*)d_in[5];
    const float* b_h2h   = (const float*)d_in[6];
    const float* w_score = (const float*)d_in[7];
    const float* W_lstm  = (const float*)d_in[8];
    const float* U_lstm  = (const float*)d_in[9];
    const float* b_lstm  = (const float*)d_in[10];

    float* out       = (float*)d_out;
    float* out_h     = out;                 // [B,H]
    float* out_c     = out + Bd * Hd;       // [B,H]
    float* out_alpha = out + 2 * Bd * Hd;   // [B,T,1]

    char*   ws    = (char*)d_ws;
    __bf16* Wswz  = (__bf16*)(ws);            // 512 KB  (32*16*32*16 bf16)
    float*  e_buf = (float*)(ws + 524288);    // 256 KB  (B*T f32)
    float*  hp    = (float*)(ws + 786432);    // 512 KB  (B*H f32)
    float*  ctx   = (float*)(ws + 1310720);   // 512 KB  (B*H f32)

    swizzle_W  <<<262144 / 256, 256, 0, stream>>>(W_i2h, Wswz);
    hp_kernel  <<<(Bd * Hd) / 256, 256, 0, stream>>>(prev_h, W_h2h, b_h2h, hp);
    attn_scores<<<(Bd * Td) / (16 * 8), 256, 0, stream>>>(batch_H, Wswz, hp, w_score, e_buf);
    softmax_ctx<<<Bd, 256, 0, stream>>>(e_buf, batch_H, out_alpha, ctx);
    lstm_kernel<<<Bd, 512, 0, stream>>>(prev_h, prev_c, onehot, ctx, W_lstm, U_lstm,
                                        b_lstm, out_h, out_c);
}
